// EvidenceGNN_89653147336994
// MI455X (gfx1250) — matrix operand order repaired and checked by reading, not compiled, not run on hardware
//
#include <hip/hip_runtime.h>
#include <hip/hip_bf16.h>
#include <math.h>

// ---------------------------------------------------------------------------
// EvidenceGNN for MI455X (gfx1250): bf16 WMMA GEMM core + fused epilogues.
// All matmuls go through v_wmma_f32_16x16x32_bf16 (K=32/instr, f32 accum).
// Each wave computes a 32x32 output tile = 4 WMMAs per K-step.
// Tile staging uses GLOBAL_LOAD_ASYNC_TO_LDS_B128 when the builtin exists.
// ---------------------------------------------------------------------------

typedef __bf16 bf16_t;
typedef __attribute__((ext_vector_type(16))) __bf16 v16bf;
typedef __attribute__((ext_vector_type(8)))  __bf16 v8bf;
typedef __attribute__((ext_vector_type(4)))  __bf16 v4bf;
typedef __attribute__((ext_vector_type(8)))  float  v8f;
typedef __attribute__((ext_vector_type(4)))  int    v4i;

#if defined(__AMDGCN__) && __has_builtin(__builtin_amdgcn_global_load_async_to_lds_b128) && __has_builtin(__builtin_amdgcn_s_wait_asynccnt)
#define USE_ASYNC 1
#define AS1 __attribute__((address_space(1)))
#define AS3 __attribute__((address_space(3)))
#else
#define USE_ASYNC 0
#endif

static constexpr int cB  = 16;
static constexpr int cN  = 256;
static constexpr int cD  = 256;
static constexpr int cH  = 8;
static constexpr int cFF = 1024;
static constexpr int cDE = 11;
static constexpr int cDA = 128;
static constexpr int cL  = 3;
static constexpr int cDK = 32;
static constexpr int cNET = 9;
static constexpr int cNP1 = cN + 1;            // 257
static constexpr int cM   = cB * cN;           // 4096 rows

static constexpr int BM = 128, BN = 64, BK = 32;

__device__ __forceinline__ float gelu_f(float x) {
  return 0.5f * x * (1.0f + erff(x * 0.70710678118654752440f));
}

// ------------------------------ small kernels ------------------------------

__global__ void cvt_bf16_kernel(const float* __restrict__ s, bf16_t* __restrict__ d, long n) {
  long i = (long)blockIdx.x * blockDim.x + threadIdx.x;
  long stride = (long)gridDim.x * blockDim.x;
  for (; i < n; i += stride) d[i] = (bf16_t)s[i];
}

__global__ __launch_bounds__(256) void embed_kernel(
    const int* __restrict__ ids, const int* __restrict__ sf,
    const float* __restrict__ edge, const float* __restrict__ symE,
    const float* __restrict__ hE, const float* __restrict__ yE,
    const float* __restrict__ rpw, const float* __restrict__ rpb,
    float* __restrict__ h) {
  int bi = blockIdx.x;          // b*N + n
  int d  = threadIdx.x;
  int id = ids[bi];
  int f0 = sf[bi * 2 + 0];
  int f1 = sf[bi * 2 + 1];
  float acc = symE[(long)id * cD + d] + hE[(long)f0 * cD + d] +
              yE[(long)f1 * cD + d] + rpb[d];
  const float* ev = edge + ((long)bi * cNP1 + cN) * cDE;   // root edge vector
#pragma unroll
  for (int e = 0; e < cDE; ++e) acc += ev[e] * rpw[e * cD + d];
  h[(long)bi * cD + d] = acc;
}

__global__ __launch_bounds__(256) void ln_kernel(
    const float* __restrict__ x, const float* __restrict__ s,
    const float* __restrict__ b, float* __restrict__ outf,
    bf16_t* __restrict__ outb) {
  __shared__ float red[256];
  int row = blockIdx.x, t = threadIdx.x;
  float v = x[(long)row * cD + t];
  red[t] = v; __syncthreads();
  for (int o = 128; o > 0; o >>= 1) { if (t < o) red[t] += red[t + o]; __syncthreads(); }
  float mean = red[0] * (1.0f / cD); __syncthreads();
  float dlt = v - mean;
  red[t] = dlt * dlt; __syncthreads();
  for (int o = 128; o > 0; o >>= 1) { if (t < o) red[t] += red[t + o]; __syncthreads(); }
  float var = red[0] * (1.0f / cD);
  float y = dlt * rsqrtf(var + 1e-5f) * s[t] + b[t];
  if (outf) outf[(long)row * cD + t] = y;
  if (outb) outb[(long)row * cD + t] = (bf16_t)y;
}

// -------------------------- generic bf16 WMMA GEMM -------------------------
// C[z] = act(A[z] @ op(W[z]) + bias[z]) (+ resid[z]); batch z = b1*Hdiv + h1.
// W normal: [K,N] row-major; wTrans: W is [N,K] row-major (i.e. B^T given).

struct GemmArgs {
  const bf16_t* A; long sAb, sAh; int lda;
  const bf16_t* W; long sWb, sWh; int ldw; int wTrans;
  const float* bias; long sBb, sBh;
  const float* resid; long sRb, sRh; int ldr;
  float* Cf; bf16_t* Cb; long sCb, sCh; int ldc;
  int M, N, K, hdiv, act;   // act: 0=none, 1=gelu
};

// A fragment: elems 0..7 = K kb..kb+7, elems 8..15 = K kb+16..kb+23 (kb in {0,8})
__device__ __forceinline__ v16bf frag_a(const bf16_t* rowp, int kb) {
  v8bf p0 = *(const v8bf*)(rowp + kb);
  v8bf p1 = *(const v8bf*)(rowp + kb + 16);
  v16bf f;
#pragma unroll
  for (int i = 0; i < 8; ++i) { f[i] = p0[i]; f[i + 8] = p1[i]; }
  return f;
}
// B fragment: elems 0..7 = K kb..kb+7, elems 8..15 = K kb+16..kb+23 (kb in {0,8}) from Bs[n][*]
__device__ __forceinline__ v16bf frag_b(const bf16_t* rowp, int kb) {
  v8bf p0 = *(const v8bf*)(rowp + kb);
  v8bf p1 = *(const v8bf*)(rowp + kb + 16);
  v16bf f;
#pragma unroll
  for (int i = 0; i < 8; ++i) { f[i] = p0[i]; f[i + 8] = p1[i]; }
  return f;
}

__device__ __forceinline__ void epi_tile(const GemmArgs& g, v8f acc, int rowBase, int colL,
                                         long biasOff, long resOff, long cOff, int lane) {
#pragma unroll
  for (int r = 0; r < 8; ++r) {
    int rowL = rowBase + r + ((lane >> 4) << 3);
    if (rowL < g.M && colL < g.N) {
      float x = acc[r];
      if (g.bias)  x += g.bias[biasOff + colL];
      if (g.act == 1) x = gelu_f(x);
      if (g.resid) x += g.resid[resOff + (long)rowL * g.ldr + colL];
      if (g.Cf) g.Cf[cOff + (long)rowL * g.ldc + colL] = x;
      if (g.Cb) g.Cb[cOff + (long)rowL * g.ldc + colL] = (bf16_t)x;
    }
  }
}

__device__ __forceinline__ void stage_b128(const bf16_t* src, bf16_t* dst) {
#if USE_ASYNC
  __builtin_amdgcn_global_load_async_to_lds_b128(
      (AS1 v4i*)(void*)src, (AS3 v4i*)(void*)dst, 0, 0);
#else
  *(v8bf*)dst = *(const v8bf*)src;
#endif
}

__global__ __launch_bounds__(256) void gemm_bf16_kernel(GemmArgs g) {
  __shared__ __align__(16) bf16_t As[BM][BK];   // 8 KB
  __shared__ __align__(16) bf16_t Bs[BN][BK];   // 4 KB, Bs[n][k]

  int z  = blockIdx.z;
  int b1 = z / g.hdiv, h1 = z % g.hdiv;
  const bf16_t* A = g.A + b1 * g.sAb + h1 * g.sAh;
  const bf16_t* W = g.W + b1 * g.sWb + h1 * g.sWh;

  int m0 = blockIdx.y * BM, n0 = blockIdx.x * BN;
  int t = threadIdx.x, lane = t & 31, w = t >> 5;
  int wm = (w & 3) * 32, wn = (w >> 2) * 32;    // 32x32 wave tile

  int l15 = lane & 15;
  int kbA = (lane < 16) ? 0 : 8;    // A fragment K base
  int kbB = (lane < 16) ? 0 : 8;    // B fragment K base

  v8f acc00, acc01, acc10, acc11;
#pragma unroll
  for (int i = 0; i < 8; ++i) { acc00[i] = 0.f; acc01[i] = 0.f; acc10[i] = 0.f; acc11[i] = 0.f; }

  // staging roles (clamped addressing; OOB rows/cols masked in epilogue)
  int arow0 = t >> 2, akg = (t & 3) << 3;       // A: slots t and t+256
  int bn = t >> 2, bkg = akg;                   // B wTrans: one slot
  int bk = t >> 3, bng = (t & 7) << 3;          // B normal: 8 scalar elems

  for (int k0 = 0; k0 < g.K; k0 += BK) {
    // ---- stage A tile: 128x32, two b128 per thread ----
    {
      int gm0 = m0 + arow0;        if (gm0 >= g.M) gm0 = g.M - 1;
      int gm1 = m0 + arow0 + 64;   if (gm1 >= g.M) gm1 = g.M - 1;
      stage_b128(A + (long)gm0 * g.lda + k0 + akg, &As[arow0][akg]);
      stage_b128(A + (long)gm1 * g.lda + k0 + akg, &As[arow0 + 64][akg]);
      if (k0 + BK < g.K)
        __builtin_prefetch((const void*)(A + (long)gm0 * g.lda + k0 + BK + akg), 0, 1);
    }
    // ---- stage B tile into Bs[n][k] ----
    if (g.wTrans) {
      int gn = n0 + bn; if (gn >= g.N) gn = g.N - 1;
      stage_b128(W + (long)gn * g.ldw + k0 + bkg, &Bs[bn][bkg]);
      if (k0 + BK < g.K)
        __builtin_prefetch((const void*)(W + (long)gn * g.ldw + k0 + BK + bkg), 0, 1);
    } else {
#pragma unroll
      for (int u = 0; u < 8; ++u) {
        int gn = n0 + bng + u; if (gn >= g.N) gn = g.N - 1;
        Bs[bng + u][bk] = W[(long)(k0 + bk) * g.ldw + gn];
      }
    }
#if USE_ASYNC
    __builtin_amdgcn_s_wait_asynccnt(0);
#endif
    __syncthreads();

    // ---- fragments per ISA 16-bit A/B layouts (wave32) ----
    v16bf a0 = frag_a(&As[wm + l15][0], kbA);
    v16bf a1 = frag_a(&As[wm + 16 + l15][0], kbA);
    v16bf b0 = frag_b(&Bs[wn + l15][0], kbB);
    v16bf b1 = frag_b(&Bs[wn + 16 + l15][0], kbB);

    acc00 = __builtin_amdgcn_wmma_f32_16x16x32_bf16(false, a0, false, b0, (short)0, acc00, false, false);
    acc01 = __builtin_amdgcn_wmma_f32_16x16x32_bf16(false, a0, false, b1, (short)0, acc01, false, false);
    acc10 = __builtin_amdgcn_wmma_f32_16x16x32_bf16(false, a1, false, b0, (short)0, acc10, false, false);
    acc11 = __builtin_amdgcn_wmma_f32_16x16x32_bf16(false, a1, false, b1, (short)0, acc11, false, false);
    __syncthreads();
  }

  long biasOff = b1 * g.sBb + h1 * g.sBh;
  long resOff  = b1 * g.sRb + h1 * g.sRh;
  long cOff    = b1 * g.sCb + h1 * g.sCh;
  int rb0 = m0 + wm, rb1 = m0 + wm + 16;
  int cl0 = n0 + wn + l15, cl1 = n0 + wn + 16 + l15;
  epi_tile(g, acc00, rb0, cl0, biasOff, resOff, cOff, lane);
  epi_tile(g, acc01, rb0, cl1, biasOff, resOff, cOff, lane);
  epi_tile(g, acc10, rb1, cl0, biasOff, resOff, cOff, lane);
  epi_tile(g, acc11, rb1, cl1, biasOff, resOff, cOff, lane);
}

// --------------------- softmax + fused edge bias + mask --------------------

__global__ __launch_bounds__(256) void softmax_bias_kernel(
    const float* __restrict__ scores, bf16_t* __restrict__ attnb,
    const float* __restrict__ edge, const float* __restrict__ We,
    const float* __restrict__ be, const unsigned char* __restrict__ pad) {
  __shared__ float red[256];
  __shared__ float wecol[12];
  int blk = blockIdx.x;               // b*H*N + h*N + i
  int i = blk % cN;
  int h = (blk / cN) % cH;
  int b = blk / (cN * cH);
  int j = threadIdx.x;
  if (j < cDE) wecol[j] = We[j * cH + h];
  __syncthreads();

  const float* ep = edge + ((long)(b * cN + i) * cNP1 + j) * cDE;
  float bias = be[h];
#pragma unroll
  for (int e = 0; e < cDE; ++e) bias += ep[e] * wecol[e];

  float s = scores[(((long)(b * cH + h) * cN) + i) * cN + j] * 0.17677669529663687f + bias;
  if (pad[b * cN + j]) s = -INFINITY;

  red[j] = s; __syncthreads();
  for (int o = 128; o > 0; o >>= 1) { if (j < o) red[j] = fmaxf(red[j], red[j + o]); __syncthreads(); }
  float mx = red[0]; __syncthreads();
  if (!isfinite(mx)) mx = 0.0f;
  float ex = __expf(s - mx);
  red[j] = ex; __syncthreads();
  for (int o = 128; o > 0; o >>= 1) { if (j < o) red[j] += red[j + o]; __syncthreads(); }
  float inv = 1.0f / fmaxf(red[0], 1e-20f);
  attnb[(((long)(b * cH + h) * cN) + i) * cN + j] = (bf16_t)(ex * inv);
}

// --------------------------- biaffine helpers ------------------------------

__global__ void rootfill_kernel(float* __restrict__ headsF, bf16_t* __restrict__ headsB,
                                const float* __restrict__ root) {
  int b = blockIdx.x, t = threadIdx.x;   // 128 threads
  long idx = ((long)b * cNP1 + cN) * cDA + t;
  headsF[idx] = root[t];
  headsB[idx] = (bf16_t)root[t];
}

__global__ __launch_bounds__(128) void hw_kernel(const float* __restrict__ headsF,
                                                 const float* __restrict__ wv,
                                                 float* __restrict__ hw) {
  __shared__ float red[128];
  int row = blockIdx.x, t = threadIdx.x;   // row = b*257 + j
  red[t] = headsF[(long)row * cDA + t] * wv[t];
  __syncthreads();
  for (int o = 64; o > 0; o >>= 1) { if (t < o) red[t] += red[t + o]; __syncthreads(); }
  if (t == 0) hw[row] = red[0];
}

__global__ void mask_kernel(float* __restrict__ out0, float* __restrict__ out2,
                            const unsigned char* __restrict__ pad) {
  long idx = (long)blockIdx.x * blockDim.x + threadIdx.x;
  long total = (long)cB * cN * cNP1;
  if (idx >= total) return;
  int j = (int)(idx % cNP1);
  int b = (int)(idx / ((long)cN * cNP1));
  if (j < cN && pad[b * cN + j]) { out0[idx] = -INFINITY; out2[idx] = -INFINITY; }
}

// ---------------------- pairwise edge-type head ----------------------------
// out[b,i,j,e] = gelu(hc[b,i,:]+hp[b,j,:]+b1)·W2[:,e] + b2[e], j<256

__global__ __launch_bounds__(256) void edge_sym_kernel(
    const float* __restrict__ hc, const float* __restrict__ hp,
    const float* __restrict__ b1, const float* __restrict__ W2,
    const float* __restrict__ b2, float* __restrict__ out) {
  __shared__ float hcs[cD];
  __shared__ float w2s[cD * cNET];
  int bi = blockIdx.x;                // b*N + i
  int b  = bi >> 8;
  int t  = threadIdx.x;
  hcs[t] = hc[(long)bi * cD + t] + b1[t];
  for (int r = t; r < cD * cNET; r += 256) w2s[r] = W2[r];
  __syncthreads();

  int jg = t >> 3, ln = t & 7;        // 32 j-slots x 8 lanes
  for (int jt = 0; jt < 8; ++jt) {
    int j = jt * 32 + jg;
    const float* hprow = hp + (long)(b * cN + j) * cD;
    float acc[cNET];
#pragma unroll
    for (int e = 0; e < cNET; ++e) acc[e] = 0.0f;
    for (int d = ln; d < cD; d += 8) {
      float g = gelu_f(hcs[d] + hprow[d]);
#pragma unroll
      for (int e = 0; e < cNET; ++e) acc[e] += g * w2s[d * cNET + e];
    }
#pragma unroll
    for (int e = 0; e < cNET; ++e) {
      acc[e] += __shfl_xor(acc[e], 4, 8);
      acc[e] += __shfl_xor(acc[e], 2, 8);
      acc[e] += __shfl_xor(acc[e], 1, 8);
    }
    if (ln == 0) {
      long base = ((long)bi * cNP1 + j) * cNET;
#pragma unroll
      for (int e = 0; e < cNET; ++e) out[base + e] = acc[e] + b2[e];
    }
  }
}

// ------------------------------- host side ---------------------------------

static void run_gemm(hipStream_t st,
                     const bf16_t* A, long sAb, long sAh, int lda,
                     const bf16_t* W, long sWb, long sWh, int ldw, int wT,
                     const float* bias, long sBb, long sBh,
                     const float* resid, long sRb, long sRh, int ldr,
                     float* Cf, bf16_t* Cb, long sCb, long sCh, int ldc,
                     int M, int N, int K, int batches, int hdiv, int act) {
  GemmArgs g;
  g.A = A; g.sAb = sAb; g.sAh = sAh; g.lda = lda;
  g.W = W; g.sWb = sWb; g.sWh = sWh; g.ldw = ldw; g.wTrans = wT;
  g.bias = bias; g.sBb = sBb; g.sBh = sBh;
  g.resid = resid; g.sRb = sRb; g.sRh = sRh; g.ldr = ldr;
  g.Cf = Cf; g.Cb = Cb; g.sCb = sCb; g.sCh = sCh; g.ldc = ldc;
  g.M = M; g.N = N; g.K = K; g.hdiv = hdiv; g.act = act;
  dim3 grid((N + BN - 1) / BN, (M + BM - 1) / BM, batches);
  gemm_bf16_kernel<<<grid, 256, 0, st>>>(g);
}

static void cvt(hipStream_t st, const float* s, bf16_t* d, long n) {
  int blocks = (int)((n + 255) / 256);
  if (blocks > 1024) blocks = 1024;
  cvt_bf16_kernel<<<blocks, 256, 0, st>>>(s, d, n);
}

extern "C" void kernel_launch(void* const* d_in, const int* in_sizes, int n_in,
                              void* d_out, int out_size, void* d_ws, size_t ws_size,
                              hipStream_t stream) {
  (void)in_sizes; (void)n_in; (void)out_size; (void)ws_size;
  // ---- inputs (setup_inputs dict order, params pytree insertion order) ----
  const int*   sym_ids = (const int*)d_in[0];
  const int*   sfeats  = (const int*)d_in[1];
  const float* edge    = (const float*)d_in[2];
  const unsigned char* pad = (const unsigned char*)d_in[3];
  auto F = [&](int i) -> const float* { return (const float*)d_in[i]; };
  const float* symE = F(4);
  const float* hE   = F(5);
  const float* yE   = F(6);
  const float* rpw  = F(7);
  const float* rpb  = F(8);
  // layers: 9 + l*18 .. ; norm at 63/64; parent 65..71; seq 72..78;
  // edge_W1 79, edge_b1 80, edge_W2 81, edge_b2 82, rt_* 83..86
  const float* norm_s = F(63);
  const float* norm_b = F(64);

  // ---- workspace bump allocator ----
  char* wsp = (char*)d_ws;
  size_t off = 0;
  auto alloc = [&](size_t bytes) -> void* {
    void* p = wsp + off;
    off += (bytes + 255) & ~(size_t)255;
    return p;
  };
  float*  h      = (float*) alloc((size_t)cM * cD * 4);
  bf16_t* xb     = (bf16_t*)alloc((size_t)cM * cD * 2);
  bf16_t* qb     = (bf16_t*)alloc((size_t)cM * cD * 2);
  bf16_t* kb     = (bf16_t*)alloc((size_t)cM * cD * 2);
  bf16_t* vb     = (bf16_t*)alloc((size_t)cM * cD * 2);
  bf16_t* aob    = (bf16_t*)alloc((size_t)cM * cD * 2);
  bf16_t* fb     = (bf16_t*)alloc((size_t)cM * cFF * 2);
  bf16_t* hb     = (bf16_t*)alloc((size_t)cM * cD * 2);
  float*  scores = (float*) alloc((size_t)cB * cH * cN * cN * 4);
  bf16_t* attnb  = (bf16_t*)alloc((size_t)cB * cH * cN * cN * 2);
  bf16_t* depb   = (bf16_t*)alloc((size_t)cM * cDA * 2);
  bf16_t* tb     = (bf16_t*)alloc((size_t)cM * cDA * 2);
  bf16_t* tab    = (bf16_t*)alloc((size_t)cM * cD * 2);
  float*  headsF = (float*) alloc((size_t)cB * cNP1 * cDA * 4);
  bf16_t* headsB = (bf16_t*)alloc((size_t)cB * cNP1 * cDA * 2);
  float*  hw     = (float*) alloc((size_t)cB * cNP1 * 4);
  float*  hc     = (float*) alloc((size_t)cM * cD * 4);
  float*  hp     = (float*) alloc((size_t)cM * cD * 4);
  bf16_t* wbA    = (bf16_t*)alloc((size_t)cD * cFF * 2);   // shared weight scratch A
  bf16_t* wbB    = (bf16_t*)alloc((size_t)cD * cFF * 2);   // shared weight scratch B
  bf16_t* ew1b   = (bf16_t*)alloc((size_t)2 * cD * cD * 2);

  float* out0 = (float*)d_out;                       // parent [B,N,257]
  float* out1 = out0 + (long)cB * cN * cNP1;         // edge types [B,N,257,9]
  float* out2 = out1 + (long)cB * cN * cNP1 * cNET;  // seq [B,N,257]

  // ---- embed ----
  embed_kernel<<<cM, 256, 0, stream>>>(sym_ids, sfeats, edge, symE, hE, yE, rpw, rpb, h);

  // ---- transformer layers ----
  for (int l = 0; l < cL; ++l) {
    int LB = 9 + l * 18;
    const float *Wq = F(LB+0), *bq = F(LB+1), *Wk = F(LB+2), *bk = F(LB+3);
    const float *Wv = F(LB+4), *bv = F(LB+5), *Wo = F(LB+6), *bo = F(LB+7);
    const float *We = F(LB+8), *be = F(LB+9);
    const float *l1s = F(LB+10), *l1b = F(LB+11);
    const float *W1 = F(LB+12), *b1 = F(LB+13), *W2 = F(LB+14), *b2 = F(LB+15);
    const float *l2s = F(LB+16), *l2b = F(LB+17);

    ln_kernel<<<cM, 256, 0, stream>>>(h, l1s, l1b, nullptr, xb);

    cvt(stream, Wq, wbA, (long)cD * cD);
    run_gemm(stream, xb,0,0,cD, wbA,0,0,cD,0, bq,0,0, nullptr,0,0,0,
             nullptr,qb,0,0,cD, cM,cD,cD, 1,1, 0);
    cvt(stream, Wk, wbA, (long)cD * cD);
    run_gemm(stream, xb,0,0,cD, wbA,0,0,cD,0, bk,0,0, nullptr,0,0,0,
             nullptr,kb,0,0,cD, cM,cD,cD, 1,1, 0);
    cvt(stream, Wv, wbA, (long)cD * cD);
    run_gemm(stream, xb,0,0,cD, wbA,0,0,cD,0, bv,0,0, nullptr,0,0,0,
             nullptr,vb,0,0,cD, cM,cD,cD, 1,1, 0);

    // scores[b,h] = Q[b,h] (256x32) @ K[b,h]^T  -> wTrans: B^T rows = K rows
    run_gemm(stream, qb,(long)cN*cD,cDK,cD,
             kb,(long)cN*cD,cDK,cD,1,
             nullptr,0,0, nullptr,0,0,0,
             scores,nullptr,(long)cH*cN*cN,(long)cN*cN,cN,
             cN,cN,cDK, cB*cH,cH, 0);

    softmax_bias_kernel<<<cB*cH*cN, 256, 0, stream>>>(scores, attnb, edge, We, be, pad);

    // out[b,h] = attn[b,h] (256x256) @ V[b,h] (256x32) -> packed back to [M,D]
    run_gemm(stream, attnb,(long)cH*cN*cN,(long)cN*cN,cN,
             vb,(long)cN*cD,cDK,cD,0,
             nullptr,0,0, nullptr,0,0,0,
             nullptr,aob,(long)cN*cD,cDK,cD,
             cN,cDK,cN, cB*cH,cH, 0);

    cvt(stream, Wo, wbA, (long)cD * cD);
    run_gemm(stream, aob,0,0,cD, wbA,0,0,cD,0, bo,0,0, h,0,0,cD,
             h,nullptr,0,0,cD, cM,cD,cD, 1,1, 0);

    ln_kernel<<<cM, 256, 0, stream>>>(h, l2s, l2b, nullptr, xb);

    cvt(stream, W1, wbA, (long)cD * cFF);
    run_gemm(stream, xb,0,0,cD, wbA,0,0,cFF,0, b1,0,0, nullptr,0,0,0,
             nullptr,fb,0,0,cFF, cM,cFF,cD, 1,1, 1);
    cvt(stream, W2, wbB, (long)cFF * cD);
    run_gemm(stream, fb,0,0,cFF, wbB,0,0,cD,0, b2,0,0, h,0,0,cD,
             h,nullptr,0,0,cD, cM,cD,cFF, 1,1, 0);
  }

  // ---- final LN ----
  ln_kernel<<<cM, 256, 0, stream>>>(h, norm_s, norm_b, nullptr, hb);

  // ---- biaffine heads (parent -> out0, seq -> out2) ----
  for (int which = 0; which < 2; ++which) {
    int PB = (which == 0) ? 65 : 72;
    const float *Wd = F(PB+0), *bd = F(PB+1), *Wh = F(PB+2), *bh = F(PB+3);
    const float *U = F(PB+4), *wv = F(PB+5), *root = F(PB+6);
    float* outp = (which == 0) ? out0 : out2;

    cvt(stream, Wd, wbA, (long)cD * cDA);
    run_gemm(stream, hb,0,0,cD, wbA,0,0,cDA,0, bd,0,0, nullptr,0,0,0,
             nullptr,depb,0,0,cDA, cM,cDA,cD, 1,1, 1);
    cvt(stream, Wh, wbA, (long)cD * cDA);
    run_gemm(stream, hb,(long)cN*cD,0,cD, wbA,0,0,cDA,0, bh,0,0, nullptr,0,0,0,
             headsF,headsB,(long)cNP1*cDA,0,cDA, cN,cDA,cD, cB,1, 1);
    rootfill_kernel<<<cB, cDA, 0, stream>>>(headsF, headsB, root);
    hw_kernel<<<cB * cNP1, cDA, 0, stream>>>(headsF, wv, hw);

    cvt(stream, U, wbA, (long)cDA * cDA);
    run_gemm(stream, depb,0,0,cDA, wbA,0,0,cDA,0, nullptr,0,0, nullptr,0,0,0,
             nullptr,tb,0,0,cDA, cM,cDA,cDA, 1,1, 0);

    // s[b] = t[b] (256x128) @ heads[b]^T (128x257) + hw[b][j]
    run_gemm(stream, tb,(long)cN*cDA,0,cDA,
             headsB,(long)cNP1*cDA,0,cDA,1,
             hw,cNP1,0, nullptr,0,0,0,
             outp,nullptr,(long)cN*cNP1,0,cNP1,
             cN,cNP1,cDA, cB,1, 0);
  }
  mask_kernel<<<(cB*cN*cNP1 + 255)/256, 256, 0, stream>>>(out0, out2, pad);

  // ---- pairwise edge-type head ----
  const float *eW1 = F(79), *eb1 = F(80), *eW2 = F(81), *eb2 = F(82);
  const float *rW1 = F(83), *rb1 = F(84), *rW2 = F(85), *rb2 = F(86);
  cvt(stream, eW1, ew1b, (long)2 * cD * cD);
  run_gemm(stream, hb,0,0,cD, ew1b,0,0,cD,0, nullptr,0,0, nullptr,0,0,0,
           hc,nullptr,0,0,cD, cM,cD,cD, 1,1, 0);
  run_gemm(stream, hb,0,0,cD, ew1b + (long)cD * cD,0,0,cD,0, nullptr,0,0, nullptr,0,0,0,
           hp,nullptr,0,0,cD, cM,cD,cD, 1,1, 0);
  edge_sym_kernel<<<cM, 256, 0, stream>>>(hc, hp, eb1, eW2, eb2, out1);

  // root column j=256 of edge types
  cvt(stream, rW1, wbA, (long)cD * cD);
  run_gemm(stream, hb,0,0,cD, wbA,0,0,cD,0, rb1,0,0, nullptr,0,0,0,
           nullptr,tab,0,0,cD, cM,cD,cD, 1,1, 1);
  cvt(stream, rW2, wbB, (long)cD * cNET);
  run_gemm(stream, tab,0,0,cD, wbB,0,0,cNET,0, rb2,0,0, nullptr,0,0,0,
           out1 + (long)cN * cNET, nullptr,0,0,cNP1 * cNET,
           cM,cNET,cD, 1,1, 0);
}
